// MultimodalRSSM_38757784879381
// MI455X (gfx1250) — matrix operand order repaired and checked by reading, compiled, not run on hardware
//
#include <hip/hip_runtime.h>

typedef __bf16 bf16_t;
typedef __attribute__((ext_vector_type(16))) __bf16 v16bf;
typedef __attribute__((ext_vector_type(8)))  float  v8f;

#define BM 128
#define BN 128
#define BK 32
#define LDS_STRIDE 40            // 80 B/row: 16B-aligned rows, gcd(20,64)=4 -> conflict-free
#define TILE_ELEMS (BM * LDS_STRIDE)

enum { EPI_BIAS = 0, EPI_RELU = 1, EPI_IMP = 2, EPI_POST = 3 };

// pack 8 f32 -> 8 bf16, single 16-byte LDS store
__device__ __forceinline__ void store8bf(bf16_t* p, float4 v0, float4 v1) {
  union { uint4 u; bf16_t h[8]; } pk;
  pk.h[0] = (bf16_t)v0.x; pk.h[1] = (bf16_t)v0.y;
  pk.h[2] = (bf16_t)v0.z; pk.h[3] = (bf16_t)v0.w;
  pk.h[4] = (bf16_t)v1.x; pk.h[5] = (bf16_t)v1.y;
  pk.h[6] = (bf16_t)v1.z; pk.h[7] = (bf16_t)v1.w;
  *(uint4*)p = pk.u;
}

// CDNA5 16-bit A fragment (16x32 MxK): lane<16 -> K {0..7,16..23}, lane>=16 -> K {8..15,24..31}
// Each half is a contiguous 8-element run -> two 16B LDS loads.
__device__ __forceinline__ v16bf frag_a_from_lds(const bf16_t* __restrict__ As, int row_base, int lane) {
  const int m  = row_base + (lane & 15);
  const int hi = (lane >> 4) & 1;
  const bf16_t* rp = As + m * LDS_STRIDE + hi * 8;
  union { uint4 u[2]; v16bf v; } r;
  r.u[0] = *(const uint4*)(rp);
  r.u[1] = *(const uint4*)(rp + 16);
  return r.v;
}

// CDNA5 16-bit B fragment (32x16 KxN): lane&15 = N, lane<16 -> K {0..7,16..23}, lane>=16 -> K {8..15,24..31}
__device__ __forceinline__ v16bf frag_b_from_lds(const bf16_t* __restrict__ Bs, int col_base, int lane) {
  const int n  = col_base + (lane & 15);
  const int k0 = ((lane >> 4) & 1) * 8;
  const bf16_t* rp = Bs + n * LDS_STRIDE + k0;
  union { uint4 u[2]; v16bf v; } r;
  r.u[0] = *(const uint4*)(rp);
  r.u[1] = *(const uint4*)(rp + 16);
  return r.v;
}

__device__ __forceinline__ void compute_chunk(const bf16_t* __restrict__ As,
                                              const bf16_t* __restrict__ Bs,
                                              int wm, int wn, int lane, v8f acc[2][4]) {
  v16bf af[2], bfr[4];
#pragma unroll
  for (int mi = 0; mi < 2; ++mi) af[mi]  = frag_a_from_lds(As, wm * 32 + mi * 16, lane);
#pragma unroll
  for (int ni = 0; ni < 4; ++ni) bfr[ni] = frag_b_from_lds(Bs, wn * 64 + ni * 16, lane);
#pragma unroll
  for (int mi = 0; mi < 2; ++mi)
#pragma unroll
    for (int ni = 0; ni < 4; ++ni)
      acc[mi][ni] = __builtin_amdgcn_wmma_f32_16x16x32_bf16(
          false, af[mi], false, bfr[ni], (short)0, acc[mi][ni], false, false);
}

// C = epi( [A1|A2] @ W^T + bias ).  W is (N, K1+K2) row-major.
// K1, K2 multiples of 32; N multiple of 128; M multiple of 16 (store tiles guarded whole).
template <int EPI>
__global__ __launch_bounds__(256)
void gemm_wmma_kernel(const float* __restrict__ A1, int lda1, int K1,
                      const float* __restrict__ A2, int lda2, int K2,
                      const float* __restrict__ W, const float* __restrict__ bias,
                      float* __restrict__ C, int ldc, int M,
                      const float* __restrict__ imp,
                      const float* __restrict__ noise, int ldn,
                      float* __restrict__ sOut, float* __restrict__ sBuf) {
  __shared__ bf16_t As[2 * TILE_ELEMS];
  __shared__ bf16_t Bs[2 * TILE_ELEMS];

  const int tid  = threadIdx.x;
  const int lane = tid & 31;
  const int w    = tid >> 5;
  const int wm   = w & 3;   // 4 waves along M (32 rows each)
  const int wn   = w >> 2;  // 2 waves along N (64 cols each)
  const int m0   = blockIdx.y * BM;
  const int n0   = blockIdx.x * BN;
  const int Ktot = K1 + K2;
  const int nk   = Ktot / BK;

  // staging: thread covers row (tid>>1), 16-col group ((tid&1)*16) for both tiles (32-bit offsets)
  const int sr = tid >> 1;
  const int sc = (tid & 1) * 16;
  const int arow  = m0 + sr;
  const int arc   = (arow < M) ? arow : (M - 1);      // clamp: rows >= M never stored
  const int aoff1 = arc * lda1 + sc;
  const int aoff2 = arc * lda2 + sc;
  const float* bp_base = W + (n0 + sr) * Ktot + sc;
  const int soff = sr * LDS_STRIDE + sc;

  v8f acc[2][4];
#pragma unroll
  for (int mi = 0; mi < 2; ++mi)
#pragma unroll
    for (int ni = 0; ni < 4; ++ni)
#pragma unroll
      for (int e = 0; e < 8; ++e) acc[mi][ni][e] = 0.0f;

  // ---- prologue: stage chunk 0 into buffer 0 ----
  {
    const float* ap = (0 < K1) ? (A1 + aoff1) : (A2 + aoff2);
    const float4 a0 = *(const float4*)(ap);
    const float4 a1 = *(const float4*)(ap + 4);
    const float4 a2 = *(const float4*)(ap + 8);
    const float4 a3 = *(const float4*)(ap + 12);
    const float4 b0 = *(const float4*)(bp_base);
    const float4 b1 = *(const float4*)(bp_base + 4);
    const float4 b2 = *(const float4*)(bp_base + 8);
    const float4 b3 = *(const float4*)(bp_base + 12);
    store8bf(As + soff,     a0, a1);
    store8bf(As + soff + 8, a2, a3);
    store8bf(Bs + soff,     b0, b1);
    store8bf(Bs + soff + 8, b2, b3);
  }
  __syncthreads();

  int buf = 0;
  for (int ki = 1; ki < nk; ++ki) {
    // issue next chunk's global loads first (overlap with WMMA below)
    const int k0 = ki * BK;
    const float* ap = (k0 < K1) ? (A1 + aoff1 + k0) : (A2 + aoff2 + (k0 - K1));
    const float* bp = bp_base + k0;
    const float4 a0 = *(const float4*)(ap);
    const float4 a1 = *(const float4*)(ap + 4);
    const float4 a2 = *(const float4*)(ap + 8);
    const float4 a3 = *(const float4*)(ap + 12);
    const float4 b0 = *(const float4*)(bp);
    const float4 b1 = *(const float4*)(bp + 4);
    const float4 b2 = *(const float4*)(bp + 8);
    const float4 b3 = *(const float4*)(bp + 12);

    compute_chunk(As + buf * TILE_ELEMS, Bs + buf * TILE_ELEMS, wm, wn, lane, acc);

    const int ob = (buf ^ 1) * TILE_ELEMS;
    store8bf(As + ob + soff,     a0, a1);
    store8bf(As + ob + soff + 8, a2, a3);
    store8bf(Bs + ob + soff,     b0, b1);
    store8bf(Bs + ob + soff + 8, b2, b3);
    __syncthreads();
    buf ^= 1;
  }
  compute_chunk(As + buf * TILE_ELEMS, Bs + buf * TILE_ELEMS, wm, wn, lane, acc);

  // ---- epilogue ----
  float w3[3] = {0.f, 0.f, 0.f};
  if (EPI == EPI_IMP) {  // softmax(importance), 3 elems
    const float i0 = imp[0], i1 = imp[1], i2 = imp[2];
    const float mx = fmaxf(i0, fmaxf(i1, i2));
    const float e0 = __expf(i0 - mx), e1 = __expf(i1 - mx), e2 = __expf(i2 - mx);
    const float inv = 1.0f / (e0 + e1 + e2);
    w3[0] = e0 * inv; w3[1] = e1 * inv; w3[2] = e2 * inv;
  }

  const int hi = (lane >> 4) & 1;
  const int nl = lane & 15;
  const int gcol0 = n0 + wn * 64 + nl;
  float bv[4];
#pragma unroll
  for (int ni = 0; ni < 4; ++ni) bv[ni] = bias[gcol0 + ni * 16];

#pragma unroll
  for (int mi = 0; mi < 2; ++mi) {
    const int rowbase = m0 + wm * 32 + mi * 16;
    if (rowbase >= M) continue;   // M % 16 == 0 at every call site -> whole tile in or out
#pragma unroll
    for (int r = 0; r < 8; ++r) {
      const int grow = rowbase + hi * 8 + r;
      if (EPI == EPI_BIAS || EPI == EPI_RELU) {
        float* crow = C + grow * ldc + gcol0;
#pragma unroll
        for (int ni = 0; ni < 4; ++ni) {
          float v = acc[mi][ni][r] + bv[ni];
          if (EPI == EPI_RELU) v = v > 0.f ? v : 0.f;
          crow[ni * 16] = v;
        }
      } else if (EPI == EPI_IMP) {
        // rows are (token*3 + modality); scatter into (token, 3*F) with importance weight
        const int tok = grow / 3, mod = grow - tok * 3;
        float* crow = C + tok * 3072 + mod * 1024 + gcol0;
        const float sc2 = w3[mod];
#pragma unroll
        for (int ni = 0; ni < 4; ++ni) crow[ni * 16] = (acc[mi][ni][r] + bv[ni]) * sc2;
      } else {  // EPI_POST
        float*       crow = C    + grow * ldc + gcol0;
        const float* nrow = noise + grow * ldn + gcol0;
        float*       srow = sOut + grow * ldc + gcol0;
        float*       brow = sBuf + grow * 1024 + gcol0;
#pragma unroll
        for (int ni = 0; ni < 4; ++ni) {
          const float v = acc[mi][ni][r] + bv[ni];
          crow[ni * 16] = v;
          const float s = v + nrow[ni * 16];
          srow[ni * 16] = s;
          brow[ni * 16] = s;
        }
      }
    }
  }
}

// One wave per (token, head): 3-token self-attention, HD=128 (4 f32 per lane)
__global__ __launch_bounds__(256)
void attn_kernel(const float* __restrict__ qkv, float* __restrict__ att) {
  const int gtid = blockIdx.x * 256 + threadIdx.x;
  const int wid  = gtid >> 5;
  const int lane = threadIdx.x & 31;
  const int n = wid >> 3;
  const int h = wid & 7;
  const long base = (long)n * 3 * 3072 + h * 128 + lane * 4;
  float q[3][4], k[3][4], v[3][4];
#pragma unroll
  for (int t = 0; t < 3; ++t) {
    const float* rp = qkv + base + (long)t * 3072;
    const float4 qv = *(const float4*)(rp);
    const float4 kv = *(const float4*)(rp + 1024);
    const float4 vv = *(const float4*)(rp + 2048);
    q[t][0] = qv.x; q[t][1] = qv.y; q[t][2] = qv.z; q[t][3] = qv.w;
    k[t][0] = kv.x; k[t][1] = kv.y; k[t][2] = kv.z; k[t][3] = kv.w;
    v[t][0] = vv.x; v[t][1] = vv.y; v[t][2] = vv.z; v[t][3] = vv.w;
  }
  float a[3][3];
#pragma unroll
  for (int qi = 0; qi < 3; ++qi)
#pragma unroll
    for (int ki = 0; ki < 3; ++ki) {
      float p = q[qi][0] * k[ki][0] + q[qi][1] * k[ki][1] +
                q[qi][2] * k[ki][2] + q[qi][3] * k[ki][3];
#pragma unroll
      for (int off = 16; off > 0; off >>= 1) p += __shfl_xor(p, off, 32);
      a[qi][ki] = p * 0.088388347648318447f;  // 1/sqrt(128)
    }
#pragma unroll
  for (int qi = 0; qi < 3; ++qi) {
    const float mx = fmaxf(a[qi][0], fmaxf(a[qi][1], a[qi][2]));
    const float e0 = __expf(a[qi][0] - mx), e1 = __expf(a[qi][1] - mx), e2 = __expf(a[qi][2] - mx);
    const float inv = 1.0f / (e0 + e1 + e2);
    a[qi][0] = e0 * inv; a[qi][1] = e1 * inv; a[qi][2] = e2 * inv;
  }
#pragma unroll
  for (int qi = 0; qi < 3; ++qi) {
    float4 o;
    o.x = a[qi][0] * v[0][0] + a[qi][1] * v[1][0] + a[qi][2] * v[2][0];
    o.y = a[qi][0] * v[0][1] + a[qi][1] * v[1][1] + a[qi][2] * v[2][1];
    o.z = a[qi][0] * v[0][2] + a[qi][1] * v[1][2] + a[qi][2] * v[2][2];
    o.w = a[qi][0] * v[0][3] + a[qi][1] * v[1][3] + a[qi][2] * v[2][3];
    *(float4*)(att + ((long)n * 3 + qi) * 1024 + h * 128 + lane * 4) = o;
  }
}

// One block per row of 1024
__global__ __launch_bounds__(256)
void layernorm_kernel(const float* __restrict__ x, const float* __restrict__ g,
                      const float* __restrict__ b, float* __restrict__ y) {
  const int row = blockIdx.x;
  const int tid = threadIdx.x;
  const float* xr = x + (long)row * 1024;
  const float4 xv = *(const float4*)(xr + tid * 4);
  float sum = xv.x + xv.y + xv.z + xv.w;
  float sq  = xv.x * xv.x + xv.y * xv.y + xv.z * xv.z + xv.w * xv.w;
#pragma unroll
  for (int off = 16; off > 0; off >>= 1) {
    sum += __shfl_xor(sum, off, 32);
    sq  += __shfl_xor(sq,  off, 32);
  }
  __shared__ float rs[8], rq[8];
  if ((tid & 31) == 0) { rs[tid >> 5] = sum; rq[tid >> 5] = sq; }
  __syncthreads();
  float ts = 0.f, tq = 0.f;
#pragma unroll
  for (int i = 0; i < 8; ++i) { ts += rs[i]; tq += rq[i]; }
  const float mean = ts * (1.0f / 1024.0f);
  const float var  = tq * (1.0f / 1024.0f) - mean * mean;
  const float inv  = rsqrtf(var + 1e-5f);
  const float4 gv = *(const float4*)(g + tid * 4);
  const float4 bv = *(const float4*)(b + tid * 4);
  float4 o;
  o.x = (xv.x - mean) * inv * gv.x + bv.x;
  o.y = (xv.y - mean) * inv * gv.y + bv.y;
  o.z = (xv.z - mean) * inv * gv.z + bv.z;
  o.w = (xv.w - mean) * inv * gv.w + bv.w;
  *(float4*)(y + (long)row * 1024 + tid * 4) = o;
}

// GRU gate combine: 64*256 elements
__global__ __launch_bounds__(256)
void gru_pointwise_kernel(const float* __restrict__ gi, const float* __restrict__ gh,
                          float* __restrict__ hbuf, float* __restrict__ outh, int ldo) {
  const int idx = blockIdx.x * 256 + threadIdx.x;
  const int b = idx >> 8, j = idx & 255;
  const float ir = gi[b * 768 + j],        hr = gh[b * 768 + j];
  const float iz = gi[b * 768 + 256 + j],  hz = gh[b * 768 + 256 + j];
  const float in_ = gi[b * 768 + 512 + j], hn = gh[b * 768 + 512 + j];
  const float r = 1.0f / (1.0f + __expf(-(ir + hr)));
  const float z = 1.0f / (1.0f + __expf(-(iz + hz)));
  const float nn = tanhf(in_ + r * hn);
  const float hp = hbuf[b * 256 + j];
  const float hv = (1.0f - z) * nn + z * hp;
  hbuf[b * 256 + j] = hv;
  outh[b * ldo + j] = hv;
}

#define GEMM(EPI, A1, LDA1, KK1, A2, LDA2, KK2, WP, BP, CP, LDC, MM, NN, IMP, NZ, LDN, SO, SB)     \
  gemm_wmma_kernel<EPI><<<dim3((NN) / BN, ((MM) + BM - 1) / BM), dim3(256), 0, stream>>>(          \
      A1, (int)(LDA1), KK1, A2, (int)(LDA2), KK2, WP, BP, CP, (int)(LDC), MM, IMP, NZ,             \
      (int)(LDN), SO, SB)

extern "C" void kernel_launch(void* const* d_in, const int* in_sizes, int n_in,
                              void* d_out, int out_size, void* d_ws, size_t ws_size,
                              hipStream_t stream) {
  (void)in_sizes; (void)n_in; (void)out_size; (void)ws_size;
  const float* emb_v = (const float*)d_in[0];
  const float* emb_t = (const float*)d_in[1];
  const float* emb_p = (const float*)d_in[2];
  const float* noise = (const float*)d_in[3];
  const float* h0    = (const float*)d_in[4];
  const float* s0    = (const float*)d_in[5];
  const float* pvw = (const float*)d_in[6];  const float* pvb = (const float*)d_in[7];
  const float* ptw = (const float*)d_in[8];  const float* ptb = (const float*)d_in[9];
  const float* ppw = (const float*)d_in[10]; const float* ppb = (const float*)d_in[11];
  const float* aiw = (const float*)d_in[12]; const float* aib = (const float*)d_in[13];
  const float* aow = (const float*)d_in[14]; const float* aob = (const float*)d_in[15];
  const float* imp = (const float*)d_in[16];
  const float* f1w = (const float*)d_in[17]; const float* f1b = (const float*)d_in[18];
  const float* f2w = (const float*)d_in[19]; const float* f2b = (const float*)d_in[20];
  const float* lng = (const float*)d_in[21]; const float* lnb = (const float*)d_in[22];
  const float* wih = (const float*)d_in[23]; const float* whh = (const float*)d_in[24];
  const float* bih = (const float*)d_in[25]; const float* bhh = (const float*)d_in[26];
  const float* prw = (const float*)d_in[27]; const float* prb = (const float*)d_in[28];
  const float* pww = (const float*)d_in[29]; const float* pwb = (const float*)d_in[30];

  float* ws  = (float*)d_ws;
  float* out = (float*)d_out;

  // workspace layout (floats), with lifetime-based aliasing
  const long O_QKV   = 25165824L;             // after stacked (24576*1024)
  const long O_FUSED = O_QKV + 75497472L;     // after qkv (24576*3072)
  const long O_GI    = O_FUSED + 8388608L;    // after fused (8192*1024)
  const long O_GH    = O_GI + 64L * 768;
  const long O_H     = O_GH + 64L * 768;
  const long O_S     = O_H + 64L * 256;

  float* stacked  = ws;                 // (24576,1024) == (8192, 3072)
  float* qkvb     = ws + O_QKV;         // (24576,3072)
  float* att      = ws;                 // aliases stacked (dead after qkv GEMM)
  float* weighted = ws + O_QKV;         // aliases qkv (dead after attn kernel)
  float* h1       = ws;                 // aliases att
  float* fpre     = ws + O_QKV;         // aliases weighted
  float* fused    = ws + O_FUSED;       // (8192,1024) persistent through RSSM
  float* gi = ws + O_GI;
  float* gh = ws + O_GH;
  float* hb = ws + O_H;
  float* sb = ws + O_S;

  const float* NF = nullptr;
  float* NW = nullptr;

  // --- phase 1: multimodal fusion (parallel over 8192 tokens) ---
  GEMM(EPI_BIAS, emb_v, 512, 512, NF, 0, 0, pvw, pvb, stacked + 0,    3072, 8192, 1024, NF, NF, 0, NW, NW);
  GEMM(EPI_BIAS, emb_t, 512, 512, NF, 0, 0, ptw, ptb, stacked + 1024, 3072, 8192, 1024, NF, NF, 0, NW, NW);
  GEMM(EPI_BIAS, emb_p, 256, 256, NF, 0, 0, ppw, ppb, stacked + 2048, 3072, 8192, 1024, NF, NF, 0, NW, NW);
  GEMM(EPI_BIAS, stacked, 1024, 1024, NF, 0, 0, aiw, aib, qkvb, 3072, 24576, 3072, NF, NF, 0, NW, NW);
  attn_kernel<<<8192, 256, 0, stream>>>(qkvb, att);
  GEMM(EPI_IMP, att, 1024, 1024, NF, 0, 0, aow, aob, weighted, 0, 24576, 1024, imp, NF, 0, NW, NW);
  GEMM(EPI_RELU, weighted, 3072, 3072, NF, 0, 0, f1w, f1b, h1, 1024, 8192, 1024, NF, NF, 0, NW, NW);
  GEMM(EPI_BIAS, h1, 1024, 1024, NF, 0, 0, f2w, f2b, fpre, 1024, 8192, 1024, NF, NF, 0, NW, NW);
  layernorm_kernel<<<8192, 256, 0, stream>>>(fpre, lng, lnb, fused);

  // --- phase 2: sequential RSSM rollout (T=128, B=64) ---
  hipMemcpyAsync(hb, h0, 64L * 256 * sizeof(float), hipMemcpyDeviceToDevice, stream);
  hipMemcpyAsync(sb, s0, 64L * 1024 * sizeof(float), hipMemcpyDeviceToDevice, stream);

  const int LDO = 128 * 3328;   // out row stride (b-major, (B,T,3328))
  const int LDF = 128 * 1024;   // fused row stride for fixed t
  for (int t = 0; t < 128; ++t) {
    const float* ft = fused + (long)t * 1024;
    // gi = [s, f_t] @ w_ih^T + b_ih ; gh = h @ w_hh^T + b_hh
    GEMM(EPI_BIAS, sb, 1024, 1024, ft, LDF, 1024, wih, bih, gi, 768, 64, 768, NF, NF, 0, NW, NW);
    GEMM(EPI_BIAS, hb, 256, 256, NF, 0, 0, whh, bhh, gh, 768, 64, 768, NF, NF, 0, NW, NW);
    gru_pointwise_kernel<<<64, 256, 0, stream>>>(gi, gh, hb, out + (long)t * 3328, LDO);
    // prior = h_new @ prior_w^T + b ; post = [h_new, f_t] @ post_w^T + b ; s_new = post + noise_t
    GEMM(EPI_BIAS, hb, 256, 256, NF, 0, 0, prw, prb, out + (long)t * 3328 + 1280, LDO, 64, 1024, NF, NF, 0, NW, NW);
    GEMM(EPI_POST, hb, 256, 256, ft, LDF, 1024, pww, pwb, out + (long)t * 3328 + 2304, LDO, 64, 1024,
         NF, noise + (long)t * 1024, 128 * 1024, out + (long)t * 3328 + 256, sb);
  }
}
